// DCNv2Optimized_41566693491553
// MI455X (gfx1250) — matrix operand order repaired and checked by reading, not compiled, not run on hardware
//
#include <hip/hip_runtime.h>

// DCNv2 for MI455X (gfx1250): two WMMA GEMM kernels with bf16 hi/lo split
// (fp32-accurate), per-tap LDS staging of im2col / bilinear-sampled panels,
// double-buffered LDS with a single barrier per tap.

typedef __attribute__((ext_vector_type(16))) __bf16 v16bf;
typedef __attribute__((ext_vector_type(8)))  float  v8f;
typedef __attribute__((ext_vector_type(4)))  unsigned v4u;
typedef __attribute__((ext_vector_type(4)))  float  v4f;

union V16 {
  v16bf bf;
  v4u   q[2];
};
static_assert(sizeof(V16) == 32, "V16 must be 32 bytes");

// Split two floats into packed bf16 hi pair and bf16 lo (residual) pair.
// Uses compiler bf16 conversions so native v_cvt_*bf16* ops are selected when
// the target has them (RNE either way).
__device__ __forceinline__ void split2(float a, float b, unsigned &hi, unsigned &lo) {
  __bf16 ha = (__bf16)a, hb = (__bf16)b;
  float ra = a - (float)ha;
  float rb = b - (float)hb;
  __bf16 la = (__bf16)ra, lb = (__bf16)rb;
  unsigned ua = __builtin_bit_cast(unsigned short, ha);
  unsigned ub = __builtin_bit_cast(unsigned short, hb);
  unsigned va = __builtin_bit_cast(unsigned short, la);
  unsigned vb = __builtin_bit_cast(unsigned short, lb);
  hi = (ub << 16) | ua;
  lo = (vb << 16) | va;
}

// ---------------------------------------------------------------------------
// Kernel 1: offset/mask conv  om[p][co] = conv3x3(x, offset_kernel) + obias
// GEMM: [128 px tile x 576] * [576 x 27(pad 32)], bf16x3 WMMA.
// ---------------------------------------------------------------------------

__device__ __forceinline__ void ofs_stage_w(const float* __restrict__ okern,
                                            int tap, int tid,
                                            unsigned* __restrict__ wh,
                                            unsigned* __restrict__ wl) {
  if (tid < 128) {
    const int s  = tid;
    const int L  = s & 31;
    const int nt = (s >> 5) & 1;
    const int ks = s >> 6;
    const int co = nt * 16 + (L & 15);
    const int coc = min(co, 26);                 // branchless: clamp + select
    const float sel = (co < 27) ? 1.f : 0.f;
    const int cibase = ks * 32 + 16 * (L >> 4);
    unsigned* dh = &wh[s * 8];
    unsigned* dl = &wl[s * 8];
#pragma unroll
    for (int j = 0; j < 8; ++j) {
      const int ci0 = cibase + 2 * j;
      float a = okern[(tap * 64 + ci0) * 27 + coc] * sel;
      float c = okern[(tap * 64 + ci0 + 1) * 27 + coc] * sel;
      unsigned hi, lo;
      split2(a, c, hi, lo);
      dh[j] = hi; dl[j] = lo;
    }
  }
}

__device__ __forceinline__ void ofs_stage_a(const float* __restrict__ x,
                                            int b, int h, int w, int c0, int p_loc,
                                            int tap,
                                            unsigned* __restrict__ ah,
                                            unsigned* __restrict__ al) {
  const int ky = tap / 3, kx = tap % 3;
  const int yy = h - 1 + ky, xx = w - 1 + kx;
  const bool valid = (yy >= 0) && (yy < 128) && (xx >= 0) && (xx < 128);
  const int cy = min(max(yy, 0), 127), cx = min(max(xx, 0), 127);
  const v4f* src = (const v4f*)(x + (((b * 128 + cy) * 128 + cx) * 64 + c0));
  const float vw = valid ? 1.f : 0.f;
  unsigned* dh = &ah[p_loc * 32 + (c0 >> 1)];
  unsigned* dl = &al[p_loc * 32 + (c0 >> 1)];
#pragma unroll
  for (int j = 0; j < 8; ++j) {
    v4f v = src[j];
    unsigned hi, lo;
    split2(v.x * vw, v.y * vw, hi, lo); dh[2 * j] = hi;     dl[2 * j] = lo;
    split2(v.z * vw, v.w * vw, hi, lo); dh[2 * j + 1] = hi; dl[2 * j + 1] = lo;
  }
}

__global__ __launch_bounds__(256) void dcn_offset_conv(
    const float* __restrict__ x,       // [4,128,128,64]
    const float* __restrict__ okern,   // [3,3,64,27]
    const float* __restrict__ obias,   // [27]
    float* __restrict__ om)            // ws: [65536][32]
{
  extern __shared__ unsigned smem[];
  unsigned* sWh = smem;                 // 2 bufs x 1024
  unsigned* sWl = smem + 2 * 1024;      // 2 bufs x 1024
  unsigned* sAh = smem + 4 * 1024;      // 2 bufs x 4096
  unsigned* sAl = smem + 4 * 1024 + 2 * 4096;

  const int tid  = threadIdx.x;
  const int lane = tid & 31;
  const int wv   = tid >> 5;           // 8 waves -> 8 M-subtiles
  const int tile = blockIdx.x * 128;   // 128 pixels per workgroup (one row)

  v8f acc[2];
  for (int nt = 0; nt < 2; ++nt)
    for (int i = 0; i < 8; ++i) acc[nt][i] = 0.f;

  const int p_loc = tid >> 1;          // pixel within tile
  const int c0    = (tid & 1) * 32;    // channel half
  const int pg    = tile + p_loc;
  const int b     = pg >> 14;
  const int h     = (pg >> 7) & 127;
  const int w     = pg & 127;
  const int row   = wv * 16 + (lane & 15);

  // prologue: stage tap 0 into buffer 0
  ofs_stage_w(okern, 0, tid, sWh, sWl);
  ofs_stage_a(x, b, h, w, c0, p_loc, 0, sAh, sAl);

  for (int tap = 0; tap < 9; ++tap) {
    const int buf = tap & 1;
    __syncthreads();
    const unsigned* wh = sWh + buf * 1024;
    const unsigned* wl = sWl + buf * 1024;
    const unsigned* ah = sAh + buf * 4096;
    const unsigned* al = sAl + buf * 4096;
    // ---- WMMA: 2 k-steps x 2 n-tiles x 3 split terms ----
#pragma unroll
    for (int ks = 0; ks < 2; ++ks) {
      V16 Ah, Al;
      const int ca = ks * 32 + 16 * (lane >> 4);  // bf16 units
      Ah.q[0] = *(const v4u*)&ah[row * 32 + (ca >> 1)];
      Ah.q[1] = *(const v4u*)&ah[row * 32 + ((ca + 8) >> 1)];
      Al.q[0] = *(const v4u*)&al[row * 32 + (ca >> 1)];
      Al.q[1] = *(const v4u*)&al[row * 32 + ((ca + 8) >> 1)];
#pragma unroll
      for (int nt = 0; nt < 2; ++nt) {
        V16 Bh, Bl;
        const int base = ((ks * 2 + nt) * 32 + lane) * 8;
        Bh.q[0] = *(const v4u*)&wh[base];
        Bh.q[1] = *(const v4u*)&wh[base + 4];
        Bl.q[0] = *(const v4u*)&wl[base];
        Bl.q[1] = *(const v4u*)&wl[base + 4];
        acc[nt] = __builtin_amdgcn_wmma_f32_16x16x32_bf16(false, Ah.bf, false, Bh.bf, (short)0, acc[nt], false, false);
        acc[nt] = __builtin_amdgcn_wmma_f32_16x16x32_bf16(false, Al.bf, false, Bh.bf, (short)0, acc[nt], false, false);
        acc[nt] = __builtin_amdgcn_wmma_f32_16x16x32_bf16(false, Ah.bf, false, Bl.bf, (short)0, acc[nt], false, false);
      }
    }
    // ---- stage next tap into the other buffer ----
    if (tap + 1 < 9) {
      const int nb = buf ^ 1;
      ofs_stage_w(okern, tap + 1, tid, sWh + nb * 1024, sWl + nb * 1024);
      ofs_stage_a(x, b, h, w, c0, p_loc, tap + 1, sAh + nb * 4096, sAl + nb * 4096);
    }
  }
  // ---- epilogue: om = acc + obias, padded stride 32 ----
#pragma unroll
  for (int nt = 0; nt < 2; ++nt) {
    const int co = nt * 16 + (lane & 15);
    const float bb = (co < 27) ? obias[min(co, 26)] : 0.f;
#pragma unroll
    for (int v = 0; v < 8; ++v) {
      const int m = v + 8 * (lane >> 4);
      const int p = tile + wv * 16 + m;
      om[(size_t)p * 32 + co] = acc[nt][v] + bb;
    }
  }
}

// ---------------------------------------------------------------------------
// Kernel 2: deformable conv  out = einsum(sampled*mask, kernel) + bias
// GEMM: [128 px tile x 576] * [576 x 64], bf16x3 WMMA, A built by bilinear
// gather per tap.
// ---------------------------------------------------------------------------

__device__ __forceinline__ void main_stage_w(const float* __restrict__ kern,
                                             int tap, int tid,
                                             unsigned* __restrict__ wh,
                                             unsigned* __restrict__ wl) {
  const int s  = tid;
  const int L  = s & 31;
  const int nt = (s >> 5) & 3;
  const int ks = s >> 7;
  const int co = nt * 16 + (L & 15);
  const int cibase = ks * 32 + 16 * (L >> 4);
  unsigned* dh = &wh[s * 8];
  unsigned* dl = &wl[s * 8];
  const float* wk = kern + (size_t)tap * 64 * 64 + co;
#pragma unroll
  for (int j = 0; j < 8; ++j) {
    const int ci0 = cibase + 2 * j;
    float a = wk[ci0 * 64];
    float c = wk[(ci0 + 1) * 64];
    unsigned hi, lo;
    split2(a, c, hi, lo);
    dh[j] = hi; dl[j] = lo;
  }
}

__device__ __forceinline__ void main_stage_a(const float* __restrict__ xb,
                                             const float* __restrict__ om,
                                             int pg, int h, int w, int c0, int p_loc,
                                             int tap,
                                             unsigned* __restrict__ ah,
                                             unsigned* __restrict__ al) {
  const int ky = tap / 3, kx = tap % 3;
  const float* omp = om + (size_t)pg * 32;
  const float dy = omp[2 * tap];
  const float dx = omp[2 * tap + 1];
  const float mk = 1.f / (1.f + __expf(-omp[18 + tap]));
  const float py = (float)(h - 1 + ky) + dy;
  const float px = (float)(w - 1 + kx) + dx;
  const float y0f = floorf(py), x0f = floorf(px);
  const float wy1 = py - y0f, wx1 = px - x0f;
  const float wy0 = 1.f - wy1, wx0 = 1.f - wx1;
  const int iy0 = (int)y0f, ix0 = (int)x0f;
  const int iy1 = iy0 + 1,  ix1 = ix0 + 1;
  const float vy0 = (iy0 >= 0 && iy0 < 128) ? 1.f : 0.f;
  const float vy1 = (iy1 >= 0 && iy1 < 128) ? 1.f : 0.f;
  const float vx0 = (ix0 >= 0 && ix0 < 128) ? 1.f : 0.f;
  const float vx1 = (ix1 >= 0 && ix1 < 128) ? 1.f : 0.f;
  const float w00 = wy0 * wx0 * vy0 * vx0 * mk;
  const float w01 = wy0 * wx1 * vy0 * vx1 * mk;
  const float w10 = wy1 * wx0 * vy1 * vx0 * mk;
  const float w11 = wy1 * wx1 * vy1 * vx1 * mk;
  const int cy0 = min(max(iy0, 0), 127), cy1 = min(max(iy1, 0), 127);
  const int cx0 = min(max(ix0, 0), 127), cx1 = min(max(ix1, 0), 127);
  const v4f* r00 = (const v4f*)(xb + ((cy0 * 128 + cx0) * 64 + c0));
  const v4f* r01 = (const v4f*)(xb + ((cy0 * 128 + cx1) * 64 + c0));
  const v4f* r10 = (const v4f*)(xb + ((cy1 * 128 + cx0) * 64 + c0));
  const v4f* r11 = (const v4f*)(xb + ((cy1 * 128 + cx1) * 64 + c0));
  unsigned* dh = &ah[p_loc * 32 + (c0 >> 1)];
  unsigned* dl = &al[p_loc * 32 + (c0 >> 1)];
#pragma unroll
  for (int j = 0; j < 8; ++j) {
    v4f a = r00[j], bq = r01[j], cq = r10[j], dq = r11[j];
    float s0 = w00 * a.x + w01 * bq.x + w10 * cq.x + w11 * dq.x;
    float s1 = w00 * a.y + w01 * bq.y + w10 * cq.y + w11 * dq.y;
    float s2 = w00 * a.z + w01 * bq.z + w10 * cq.z + w11 * dq.z;
    float s3 = w00 * a.w + w01 * bq.w + w10 * cq.w + w11 * dq.w;
    unsigned hi, lo;
    split2(s0, s1, hi, lo); dh[2 * j] = hi;     dl[2 * j] = lo;
    split2(s2, s3, hi, lo); dh[2 * j + 1] = hi; dl[2 * j + 1] = lo;
  }
}

__global__ __launch_bounds__(256) void dcn_main(
    const float* __restrict__ x,     // [4,128,128,64]
    const float* __restrict__ kern,  // [3,3,64,64]
    const float* __restrict__ bias,  // [64]
    const float* __restrict__ om,    // ws: [65536][32]
    float* __restrict__ out)         // [65536][64]
{
  extern __shared__ unsigned smem[];
  unsigned* sWh = smem;                 // 2 bufs x 2048
  unsigned* sWl = smem + 2 * 2048;      // 2 bufs x 2048
  unsigned* sAh = smem + 4 * 2048;      // 2 bufs x 4096
  unsigned* sAl = smem + 4 * 2048 + 2 * 4096;

  const int tid  = threadIdx.x;
  const int lane = tid & 31;
  const int wv   = tid >> 5;
  const int tile = blockIdx.x * 128;

  v8f acc[4];
  for (int nt = 0; nt < 4; ++nt)
    for (int i = 0; i < 8; ++i) acc[nt][i] = 0.f;

  const int p_loc = tid >> 1;
  const int c0    = (tid & 1) * 32;
  const int pg    = tile + p_loc;
  const int b     = pg >> 14;
  const int h     = (pg >> 7) & 127;
  const int w     = pg & 127;
  const float* xb = x + (size_t)b * (128 * 128 * 64);
  const int row   = wv * 16 + (lane & 15);

  // prologue: stage tap 0 into buffer 0
  main_stage_w(kern, 0, tid, sWh, sWl);
  main_stage_a(xb, om, pg, h, w, c0, p_loc, 0, sAh, sAl);

  for (int tap = 0; tap < 9; ++tap) {
    const int buf = tap & 1;
    __syncthreads();
    const unsigned* wh = sWh + buf * 2048;
    const unsigned* wl = sWl + buf * 2048;
    const unsigned* ah = sAh + buf * 4096;
    const unsigned* al = sAl + buf * 4096;
    // ---- WMMA: 2 k-steps x 4 n-tiles x 3 split terms ----
#pragma unroll
    for (int ks = 0; ks < 2; ++ks) {
      V16 Ah, Al;
      const int ca = ks * 32 + 16 * (lane >> 4);
      Ah.q[0] = *(const v4u*)&ah[row * 32 + (ca >> 1)];
      Ah.q[1] = *(const v4u*)&ah[row * 32 + ((ca + 8) >> 1)];
      Al.q[0] = *(const v4u*)&al[row * 32 + (ca >> 1)];
      Al.q[1] = *(const v4u*)&al[row * 32 + ((ca + 8) >> 1)];
#pragma unroll
      for (int nt = 0; nt < 4; ++nt) {
        V16 Bh, Bl;
        const int base = ((ks * 4 + nt) * 32 + lane) * 8;
        Bh.q[0] = *(const v4u*)&wh[base];
        Bh.q[1] = *(const v4u*)&wh[base + 4];
        Bl.q[0] = *(const v4u*)&wl[base];
        Bl.q[1] = *(const v4u*)&wl[base + 4];
        acc[nt] = __builtin_amdgcn_wmma_f32_16x16x32_bf16(false, Ah.bf, false, Bh.bf, (short)0, acc[nt], false, false);
        acc[nt] = __builtin_amdgcn_wmma_f32_16x16x32_bf16(false, Al.bf, false, Bh.bf, (short)0, acc[nt], false, false);
        acc[nt] = __builtin_amdgcn_wmma_f32_16x16x32_bf16(false, Ah.bf, false, Bl.bf, (short)0, acc[nt], false, false);
      }
    }
    // ---- stage next tap into the other buffer ----
    if (tap + 1 < 9) {
      const int nb = buf ^ 1;
      main_stage_w(kern, tap + 1, tid, sWh + nb * 2048, sWl + nb * 2048);
      main_stage_a(xb, om, pg, h, w, c0, p_loc, tap + 1, sAh + nb * 4096, sAl + nb * 4096);
    }
  }
  // ---- epilogue: add bias, store ----
#pragma unroll
  for (int nt = 0; nt < 4; ++nt) {
    const int co = nt * 16 + (lane & 15);
    const float bb = bias[co];
#pragma unroll
    for (int v = 0; v < 8; ++v) {
      const int m = v + 8 * (lane >> 4);
      const int p = tile + wv * 16 + m;
      out[(size_t)p * 64 + co] = acc[nt][v] + bb;
    }
  }
}

// ---------------------------------------------------------------------------
extern "C" void kernel_launch(void* const* d_in, const int* in_sizes, int n_in,
                              void* d_out, int out_size, void* d_ws, size_t ws_size,
                              hipStream_t stream) {
  (void)in_sizes; (void)n_in; (void)out_size; (void)ws_size;
  const float* x     = (const float*)d_in[0];  // [4,128,128,64]
  const float* kern  = (const float*)d_in[1];  // [3,3,64,64]
  const float* bias  = (const float*)d_in[2];  // [64]
  const float* okern = (const float*)d_in[3];  // [3,3,64,27]
  const float* obias = (const float*)d_in[4];  // [27]
  float* out = (float*)d_out;                  // [4,128,128,64]
  float* om  = (float*)d_ws;                   // [65536][32] = 8 MB

  const size_t shmem1 = (size_t)(2 * 1024 + 2 * 1024 + 2 * 4096 + 2 * 4096) * 4;  // 80 KB
  const size_t shmem2 = (size_t)(2 * 2048 + 2 * 2048 + 2 * 4096 + 2 * 4096) * 4;  // 96 KB

  dcn_offset_conv<<<512, 256, shmem1, stream>>>(x, okern, obias, om);
  dcn_main<<<512, 256, shmem2, stream>>>(x, kern, bias, om, out);
}
